// LFISTA_40913858462097
// MI455X (gfx1250) — matrix operand order repaired and checked by reading, not compiled, not run on hardware
//
#include <hip/hip_runtime.h>
#include <math.h>

#define S_DIM 2048
#define B_DIM 1024
#define FISTA_ITERS 16
#define POW_ITERS 40

typedef __bf16 bf16_t;
typedef __attribute__((ext_vector_type(8)))  bf16_t v8bf;
typedef __attribute__((ext_vector_type(16))) bf16_t v16bf;
typedef __attribute__((ext_vector_type(8)))  float  v8f;

union BF16x16 { v16bf v; v8bf h[2]; };
union F32x8   { v8f v; float f[8]; };

// ---------------- GEMM tiling ----------------
#define BM 128
#define BN 64
#define BK 32
#define KSTAGES (S_DIM / BK)                 // 64
#define LDSA 40                              // padded row stride (bf16)
#define LDSB 40
#define AH_OFF 0
#define AL_OFF (BM * LDSA)                   // 5120
#define BH_OFF (2 * BM * LDSA)               // 10240
#define BL_OFF (2 * BM * LDSA + BN * LDSB)   // 12800
#define STAGE_SZ (2 * BM * LDSA + 2 * BN * LDSB) // 15360 bf16 = 30720 B

// CDNA5 async global->LDS copy (ASYNCcnt-tracked), bypasses VGPRs entirely.
__device__ __forceinline__ void async_ld_b128(const bf16_t* lds_dst, const void* gaddr)
{
  // Low 32 bits of the generic LDS pointer are the LDS byte offset
  // (per the ISA's LDS aperture rule).
  unsigned lds_off = (unsigned)(uintptr_t)lds_dst;
  asm volatile("global_load_async_to_lds_b128 %0, %1, off"
               :: "v"(lds_off), "v"(gaddr)
               : "memory");
}
__device__ __forceinline__ void wait_async0()
{
  asm volatile("s_wait_asynccnt 0" ::: "memory");
}

// Split-bf16 mainloop: acc += Ah*Bh + Ah*Bl + Al*Bh over full K.
// A matrices: [S][S] row-major. B matrices: [N=1024][K=S] row-major.
__device__ __forceinline__ void gemm_mainloop(
    const bf16_t* __restrict__ Ahg, const bf16_t* __restrict__ Alg,
    const bf16_t* __restrict__ Bhg, const bf16_t* __restrict__ Blg,
    int m0, int n0, bf16_t* lds, v8f (&acc)[2][2])
{
  const int t    = threadIdx.x;
  const int wave = t >> 5;
  const int lane = t & 31;
  const int wm   = wave >> 1;      // 0..3
  const int wn   = wave & 1;       // 0..1
  const int half = lane >> 4;
  const int lm   = lane & 15;

  // global tile load mapping: 16B (8 x bf16) segments
  const int ar = t >> 2;           // 0..63
  const int ac = (t & 3) * 8;      // 0,8,16,24
  const size_t agA0 = (size_t)(m0 + ar)      * S_DIM + ac;
  const size_t agA1 = (size_t)(m0 + 64 + ar) * S_DIM + ac;
  const size_t agB  = (size_t)(n0 + ar)      * S_DIM + ac;
  const int sA0 = ar * LDSA + ac;
  const int sA1 = (64 + ar) * LDSA + ac;
  const int sB  = ar * LDSB + ac;

  // stage 0 -> buffer 0 (async DMA into LDS)
  {
    bf16_t* s = lds;
    async_ld_b128(s + AH_OFF + sA0, Ahg + agA0);
    async_ld_b128(s + AH_OFF + sA1, Ahg + agA1);
    async_ld_b128(s + AL_OFF + sA0, Alg + agA0);
    async_ld_b128(s + AL_OFF + sA1, Alg + agA1);
    async_ld_b128(s + BH_OFF + sB,  Bhg + agB);
    async_ld_b128(s + BL_OFF + sB,  Blg + agB);
  }
  wait_async0();
  __syncthreads();

  for (int kk = 0; kk < KSTAGES; ++kk) {
    const int nxt = kk + 1;
    if (nxt < KSTAGES) {
      // prefetch next stage into the other buffer (last read 2 stages ago,
      // protected by the barrier at the end of the previous iteration)
      const size_t ko = (size_t)nxt * BK;
      bf16_t* s = lds + (size_t)(nxt & 1) * STAGE_SZ;
      async_ld_b128(s + AH_OFF + sA0, Ahg + agA0 + ko);
      async_ld_b128(s + AH_OFF + sA1, Ahg + agA1 + ko);
      async_ld_b128(s + AL_OFF + sA0, Alg + agA0 + ko);
      async_ld_b128(s + AL_OFF + sA1, Alg + agA1 + ko);
      async_ld_b128(s + BH_OFF + sB,  Bhg + agB + ko);
      async_ld_b128(s + BL_OFF + sB,  Blg + agB + ko);
    }
    {
      const bf16_t* s = lds + (size_t)(kk & 1) * STAGE_SZ;
      const int q0 = half * 8;   // A fragment: K = {q0..q0+7} U {q0+16..q0+23}
      BF16x16 ah[2], al[2], bh[2], bl[2];
#pragma unroll
      for (int tm = 0; tm < 2; ++tm) {
        const bf16_t* pa = s + AH_OFF + (wm * 32 + tm * 16 + lm) * LDSA;
        ah[tm].h[0] = *(const v8bf*)(pa + q0);
        ah[tm].h[1] = *(const v8bf*)(pa + q0 + 16);
        const bf16_t* pl = s + AL_OFF + (wm * 32 + tm * 16 + lm) * LDSA;
        al[tm].h[0] = *(const v8bf*)(pl + q0);
        al[tm].h[1] = *(const v8bf*)(pl + q0 + 16);
      }
#pragma unroll
      for (int tn = 0; tn < 2; ++tn) {   // B fragment: K = {q0..q0+7} U {q0+16..q0+23}
        const bf16_t* pb = s + BH_OFF + (wn * 32 + tn * 16 + lm) * LDSB + half * 8;
        bh[tn].h[0] = *(const v8bf*)(pb);
        bh[tn].h[1] = *(const v8bf*)(pb + 16);
        const bf16_t* pc = s + BL_OFF + (wn * 32 + tn * 16 + lm) * LDSB + half * 8;
        bl[tn].h[0] = *(const v8bf*)(pc);
        bl[tn].h[1] = *(const v8bf*)(pc + 16);
      }
#pragma unroll
      for (int tm = 0; tm < 2; ++tm)
#pragma unroll
        for (int tn = 0; tn < 2; ++tn) {
          acc[tm][tn] = __builtin_amdgcn_wmma_f32_16x16x32_bf16(
              false, ah[tm].v, false, bh[tn].v, (short)0, acc[tm][tn], false, false);
          acc[tm][tn] = __builtin_amdgcn_wmma_f32_16x16x32_bf16(
              false, ah[tm].v, false, bl[tn].v, (short)0, acc[tm][tn], false, false);
          acc[tm][tn] = __builtin_amdgcn_wmma_f32_16x16x32_bf16(
              false, al[tm].v, false, bh[tn].v, (short)0, acc[tm][tn], false, false);
        }
    }
    if (nxt < KSTAGES) wait_async0();  // our async DMAs for stage nxt landed
    __syncthreads();                   // everyone done reading kk & DMAs visible
  }
}

// GEMM1: P = W @ yθ ; epilogue: res = Y - (src*P + yδ); g = src*res (bf16 split)
__global__ __launch_bounds__(256) void lfista_gemm1(
    const bf16_t* __restrict__ Whi, const bf16_t* __restrict__ Wlo,
    const bf16_t* __restrict__ Bhi, const bf16_t* __restrict__ Blo,
    const float* __restrict__ srcB, const float* __restrict__ YB,
    const float* __restrict__ ybuf,
    float* __restrict__ rbuf, bf16_t* __restrict__ Ghi, bf16_t* __restrict__ Glo)
{
  __shared__ bf16_t lds[2 * STAGE_SZ];
  const int n0 = blockIdx.x * BN;
  const int m0 = blockIdx.y * BM;
  v8f acc[2][2] = {};
  gemm_mainloop(Whi, Wlo, Bhi, Blo, m0, n0, lds, acc);

  const int t = threadIdx.x, wave = t >> 5, lane = t & 31;
  const int wm = wave >> 1, wn = wave & 1, half = lane >> 4, lm = lane & 15;
#pragma unroll
  for (int tm = 0; tm < 2; ++tm)
#pragma unroll
    for (int tn = 0; tn < 2; ++tn) {
      const int b  = n0 + wn * 32 + tn * 16 + lm;
      const int s0 = m0 + wm * 32 + tm * 16 + half * 8;
      const size_t pS  = (size_t)b * S_DIM + s0;
      const size_t pYd = (size_t)b * (2 * S_DIM) + S_DIM + s0;
      F32x8 a; a.v = acc[tm][tn];
#pragma unroll
      for (int i = 0; i < 8; ++i) {
        const float sv  = srcB[pS + i];
        const float res = YB[pS + i] - (sv * a.f[i] + ybuf[pYd + i]);
        rbuf[pS + i] = res;
        const float g  = sv * res;
        const bf16_t gh = (bf16_t)g;
        Ghi[pS + i] = gh;
        Glo[pS + i] = (bf16_t)(g - (float)gh);
      }
    }
}

// GEMM2: Q = Wᵀ @ g ; epilogue: vθ = yθ + Q/L; x = soft(v); y = x + c(x - x_old)
__global__ __launch_bounds__(256) void lfista_gemm2(
    const bf16_t* __restrict__ WThi, const bf16_t* __restrict__ WTlo,
    const bf16_t* __restrict__ Ghi, const bf16_t* __restrict__ Glo,
    float* __restrict__ ybuf, float* __restrict__ xbuf,
    bf16_t* __restrict__ Bhi, bf16_t* __restrict__ Blo,
    const float* __restrict__ scal, float cmix)
{
  __shared__ bf16_t lds[2 * STAGE_SZ];
  const int n0 = blockIdx.x * BN;
  const int m0 = blockIdx.y * BM;
  v8f acc[2][2] = {};
  gemm_mainloop(WThi, WTlo, Ghi, Glo, m0, n0, lds, acc);

  const float invL = scal[1];
  const float thr  = scal[2];
  const int t = threadIdx.x, wave = t >> 5, lane = t & 31;
  const int wm = wave >> 1, wn = wave & 1, half = lane >> 4, lm = lane & 15;
#pragma unroll
  for (int tm = 0; tm < 2; ++tm)
#pragma unroll
    for (int tn = 0; tn < 2; ++tn) {
      const int b  = n0 + wn * 32 + tn * 16 + lm;
      const int s0 = m0 + wm * 32 + tm * 16 + half * 8;
      const size_t pS = (size_t)b * S_DIM + s0;
      const size_t pX = (size_t)b * (2 * S_DIM) + s0;
      F32x8 a; a.v = acc[tm][tn];
#pragma unroll
      for (int i = 0; i < 8; ++i) {
        const float v  = ybuf[pX + i] + a.f[i] * invL;
        float xn = fmaxf(fabsf(v) - thr, 0.0f);
        xn = copysignf(xn, v);
        const float xo = xbuf[pX + i];
        const float yn = xn + cmix * (xn - xo);
        xbuf[pX + i] = xn;
        ybuf[pX + i] = yn;
        const bf16_t h = (bf16_t)yn;
        Bhi[pS + i] = h;
        Blo[pS + i] = (bf16_t)(yn - (float)h);
      }
    }
}

// delta half: vδ = yδ + res/L; soft-threshold; momentum
__global__ __launch_bounds__(256) void lfista_delta(
    const float* __restrict__ rbuf, float* __restrict__ xbuf,
    float* __restrict__ ybuf, const float* __restrict__ scal, float cmix)
{
  const int idx = blockIdx.x * 256 + threadIdx.x;     // [0, B*S)
  const int b = idx >> 11, s = idx & (S_DIM - 1);
  const float invL = scal[1];
  const float thr  = scal[2];
  const size_t pX = (size_t)b * (2 * S_DIM) + S_DIM + s;
  const float v = ybuf[pX] + rbuf[idx] * invL;
  float xn = fmaxf(fabsf(v) - thr, 0.0f);
  xn = copysignf(xn, v);
  const float xo = xbuf[pX];
  xbuf[pX] = xn;
  ybuf[pX] = xn + cmix * (xn - xo);
}

// one-time: split W into bf16 hi/lo + transposed copies
__global__ __launch_bounds__(256) void lfista_splitw(
    const float* __restrict__ W, bf16_t* __restrict__ Whi, bf16_t* __restrict__ Wlo,
    bf16_t* __restrict__ WThi, bf16_t* __restrict__ WTlo)
{
  const int idx = blockIdx.x * 256 + threadIdx.x;     // [0, S*S)
  const int i = idx >> 11, j = idx & (S_DIM - 1);
  const float w = W[idx];
  const bf16_t h = (bf16_t)w;
  const bf16_t l = (bf16_t)(w - (float)h);
  Whi[idx] = h; Wlo[idx] = l;
  const size_t tidx = (size_t)j * S_DIM + i;
  WThi[tidx] = h; WTlo[tidx] = l;
}

__global__ __launch_bounds__(256) void lfista_zero(float* __restrict__ p, int n)
{
  const int i = blockIdx.x * 256 + threadIdx.x;
  if (i < n) p[i] = 0.0f;
}

// ---- power iteration for L = lambda_max(WᵀW) ----
__global__ __launch_bounds__(256) void lfista_pow_init(float* v, float* scal)
{
  const int i = blockIdx.x * 256 + threadIdx.x;
  if (i < S_DIM) v[i] = 1.0f;
  if (i < 8) scal[i] = 0.0f;
}

__global__ __launch_bounds__(256) void lfista_matvec(
    const float* __restrict__ W, const float* __restrict__ v, float* __restrict__ u)
{
  __shared__ float red[256];
  const int row = blockIdx.x, t = threadIdx.x;
  float s = 0.0f;
  for (int k = t; k < S_DIM; k += 256) s += W[(size_t)row * S_DIM + k] * v[k];
  red[t] = s; __syncthreads();
  for (int o = 128; o > 0; o >>= 1) { if (t < o) red[t] += red[t + o]; __syncthreads(); }
  if (!t) u[row] = red[0];
}

__global__ __launch_bounds__(256) void lfista_matvecT(
    const float* __restrict__ W, const float* __restrict__ u,
    float* __restrict__ vp, float* __restrict__ scal)
{
  __shared__ float red[256];
  const int col = blockIdx.x, t = threadIdx.x;
  float s = 0.0f;
  for (int k = t; k < S_DIM; k += 256) s += W[(size_t)k * S_DIM + col] * u[k];
  red[t] = s; __syncthreads();
  for (int o = 128; o > 0; o >>= 1) { if (t < o) red[t] += red[t + o]; __syncthreads(); }
  if (!t) { vp[col] = red[0]; atomicAdd(&scal[3], red[0] * red[0]); }
}

__global__ __launch_bounds__(256) void lfista_pow_fin(
    float* __restrict__ v, const float* __restrict__ vp,
    float* __restrict__ scal, const float* __restrict__ alpha)
{
  __shared__ float l2s;
  const int t = threadIdx.x;
  if (!t) l2s = scal[3];
  __syncthreads();
  const float L  = sqrtf(l2s);        // ||WᵀW v|| with ||v||=1  ->  lambda_max
  const float rL = 1.0f / L;
  for (int i = t; i < S_DIM; i += 256) v[i] = vp[i] * rL;
  if (!t) { scal[0] = L; scal[1] = rL; scal[2] = 0.5f * alpha[0] * rL; scal[3] = 0.0f; }
}

extern "C" void kernel_launch(void* const* d_in, const int* in_sizes, int n_in,
                              void* d_out, int out_size, void* d_ws, size_t ws_size,
                              hipStream_t stream)
{
  (void)in_sizes; (void)n_in; (void)out_size; (void)ws_size;
  const float* srcB  = (const float*)d_in[0];   // [B][S]
  const float* YB    = (const float*)d_in[1];   // [B][S]
  const float* W     = (const float*)d_in[2];   // [S][S]
  const float* alpha = (const float*)d_in[3];   // [1]

  char* ws = (char*)d_ws;
  const size_t SZ_WBF = (size_t)S_DIM * S_DIM * sizeof(bf16_t);      // 8 MB
  const size_t SZ_ABF = (size_t)B_DIM * S_DIM * sizeof(bf16_t);      // 4 MB
  const size_t SZ_RF  = (size_t)B_DIM * S_DIM * sizeof(float);       // 8 MB
  const size_t SZ_XF  = (size_t)B_DIM * 2 * S_DIM * sizeof(float);   // 16 MB
  size_t off = 0;
  bf16_t* Whi  = (bf16_t*)(ws + off); off += SZ_WBF;
  bf16_t* Wlo  = (bf16_t*)(ws + off); off += SZ_WBF;
  bf16_t* WThi = (bf16_t*)(ws + off); off += SZ_WBF;
  bf16_t* WTlo = (bf16_t*)(ws + off); off += SZ_WBF;
  bf16_t* Bhi  = (bf16_t*)(ws + off); off += SZ_ABF;
  bf16_t* Blo  = (bf16_t*)(ws + off); off += SZ_ABF;
  bf16_t* Ghi  = (bf16_t*)(ws + off); off += SZ_ABF;
  bf16_t* Glo  = (bf16_t*)(ws + off); off += SZ_ABF;
  float*  rbuf = (float*)(ws + off);  off += SZ_RF;
  float*  xbuf = (float*)(ws + off);  off += SZ_XF;
  float*  ybuf = (float*)(ws + off);  off += SZ_XF;   // contiguous after xbuf
  float*  pv   = (float*)(ws + off);  off += S_DIM * sizeof(float);
  float*  pu   = (float*)(ws + off);  off += S_DIM * sizeof(float);
  float*  pvp  = (float*)(ws + off);  off += S_DIM * sizeof(float);
  float*  scal = (float*)(ws + off);  off += 64;

  // FISTA momentum coefficients (data-independent scalar recursion)
  float cmix[FISTA_ITERS];
  {
    float t = 1.0f;
    for (int k = 0; k < FISTA_ITERS; ++k) {
      const float tn = (1.0f + sqrtf(1.0f + 4.0f * t * t)) * 0.5f;
      cmix[k] = (t - 1.0f) / tn;
      t = tn;
    }
  }

  // 1) split W into bf16 hi/lo (+ transposed copies for Wᵀ GEMM)
  lfista_splitw<<<(S_DIM * S_DIM) / 256, 256, 0, stream>>>(W, Whi, Wlo, WThi, WTlo);

  // 2) zero state: x|y contiguous (8M floats), yθ bf16 hi|lo contiguous (2M "floats")
  lfista_zero<<<(int)((2 * SZ_XF / sizeof(float)) / 256), 256, 0, stream>>>(
      xbuf, (int)(2 * SZ_XF / sizeof(float)));
  lfista_zero<<<(int)((2 * SZ_ABF / sizeof(float)) / 256), 256, 0, stream>>>(
      (float*)Bhi, (int)(2 * SZ_ABF / sizeof(float)));

  // 3) L = lambda_max(WᵀW) via power iteration
  lfista_pow_init<<<8, 256, 0, stream>>>(pv, scal);
  for (int it = 0; it < POW_ITERS; ++it) {
    lfista_matvec <<<S_DIM, 256, 0, stream>>>(W, pv, pu);
    lfista_matvecT<<<S_DIM, 256, 0, stream>>>(W, pu, pvp, scal);
    lfista_pow_fin<<<1, 256, 0, stream>>>(pv, pvp, scal, alpha);
  }

  // 4) FISTA main loop: 2 split-bf16 WMMA GEMMs + fused elementwise per iteration
  dim3 gg(B_DIM / BN, S_DIM / BM);   // (16, 16)
  for (int k = 0; k < FISTA_ITERS; ++k) {
    lfista_gemm1<<<gg, 256, 0, stream>>>(Whi, Wlo, Bhi, Blo, srcB, YB, ybuf,
                                         rbuf, Ghi, Glo);
    lfista_gemm2<<<gg, 256, 0, stream>>>(WThi, WTlo, Ghi, Glo, ybuf, xbuf,
                                         Bhi, Blo, scal, cmix[k]);
    lfista_delta<<<(B_DIM * S_DIM) / 256, 256, 0, stream>>>(rbuf, xbuf, ybuf,
                                                            scal, cmix[k]);
  }

  // 5) output x (already in (B, 2S) layout)
  hipMemcpyAsync(d_out, xbuf, SZ_XF, hipMemcpyDeviceToDevice, stream);
}
